// STGCN_wo_NL_26353919328691
// MI455X (gfx1250) — matrix operand order repaired and checked by reading, not compiled, not run on hardware
//
#include <hip/hip_runtime.h>
#include <hip/hip_bf16.h>

typedef _Float16 hlf;
typedef _Float16 v16h __attribute__((ext_vector_type(16)));
typedef _Float16 v8h  __attribute__((ext_vector_type(8)));
typedef float    v8f  __attribute__((ext_vector_type(8)));

// Problem constants
static constexpr int B_  = 128;
static constexpr int S_  = 256;
static constexpr int H_  = 256;
static constexpr long BS = (long)B_ * S_;   // 32768

// ---------------------------------------------------------------------------
// WMMA fragment loaders (CDNA5 v_wmma_f32_16x16x32_f16, wave32)
// A: 16x32 (MxK). lane<16: M=lane, K in {k0..k0+7, k0+16..k0+23}
//                 lane>=16: M=lane-16, K in {k0+8..k0+15, k0+24..k0+31}
// ---------------------------------------------------------------------------
__device__ __forceinline__ v16h frag_a_load(const hlf* A, int lda, int lane, int k0) {
  int hi = lane >> 4;
  int r  = lane & 15;
  const hlf* p = A + (long)r * lda + k0 + hi * 8;
  v8h lo = *(const v8h*)p;          // K = k0+8*hi .. +7
  v8h hh = *(const v8h*)(p + 16);   // K = k0+16+8*hi .. +7
  v16h out;
#pragma unroll
  for (int i = 0; i < 8; ++i) { out[i] = lo[i]; out[8 + i] = hh[i]; }
  return out;
}

// B stored [N,K] row-major: lane c=lane&15 -> column n0+c,
// K in {k0+8*hi..+7, k0+16+8*hi..+7}, hi = lane>>4.
__device__ __forceinline__ v16h frag_b_nk(const hlf* Bt, int ldbt, int n0, int lane, int k0) {
  int hi = lane >> 4;
  int c  = lane & 15;
  const hlf* p = Bt + (long)(n0 + c) * ldbt + k0 + hi * 8;
  v8h lo = *(const v8h*)p;
  v8h hh = *(const v8h*)(p + 16);
  v16h out;
#pragma unroll
  for (int i = 0; i < 8; ++i) { out[i] = lo[i]; out[8 + i] = hh[i]; }
  return out;
}

// ---------------------------------------------------------------------------
// LDS-tiled WMMA GEMM, [N,K]-layout B operand for all cases.
// Workgroup = 256 threads (8 waves): 128M x 64N tile; each wave 16M x 64N.
// B tile (64N x 32K) staged via GLOBAL_LOAD_ASYNC_TO_LDS_B128, double-buffered
// and pipelined with s_wait_asynccnt (ASYNCcnt).
// ct=1 writes the f16/f32 outputs transposed per batch: C[n, m].
// ---------------------------------------------------------------------------
__global__ void __launch_bounds__(256)
wmma_gemm_lds(const hlf* A, long sA, int lda,
              const hlf* Bt, long sB, int ldb,
              const float* bias, float alpha,
              float* Cf, long sCf, int ldcf,
              hlf* Ch, long sCh, int ldch,
              int M, int N, int K, int ct)
{
  __shared__ hlf sBt[2][64 * 32];   // [buf][n_local*32 + k_local]

  int tid  = threadIdx.x;
  int w    = tid >> 5;
  int lane = tid & 31;
  int m0     = blockIdx.y * 128 + w * 16;
  int n0base = blockIdx.x * 64;
  int z      = blockIdx.z;
  const hlf* Ab = A + (long)z * sA + (long)m0 * lda;
  const hlf* Bb = Bt + (long)z * sB;

  // async-copy assignment: thread -> (row n, 16-byte quarter q) of the B tile
  int bn = tid >> 2, bq = tid & 3;
  const hlf* bsrc = Bb + (long)(n0base + bn) * ldb + bq * 8;
  unsigned bdst0 = (unsigned)(unsigned long long)&sBt[0][bn * 32 + bq * 8];
  unsigned bdst1 = (unsigned)(unsigned long long)&sBt[1][bn * 32 + bq * 8];

  // stage k-tile 0 into buffer 0
  asm volatile("global_load_async_to_lds_b128 %0, %1, off"
               :: "v"(bdst0), "v"(bsrc) : "memory");

  v8f acc[4] = {};
  int nk = K >> 5;
  for (int ks = 0; ks < nk; ++ks) {
    int k0 = ks << 5;
    if (ks + 1 < nk) {
      unsigned nb = (ks & 1) ? bdst0 : bdst1;
      asm volatile("global_load_async_to_lds_b128 %0, %1, off"
                   :: "v"(nb), "v"(bsrc + k0 + 32) : "memory");
      // oldest (current tile) done; newest may still be in flight
      asm volatile("s_wait_asynccnt 0x1" ::: "memory");
    } else {
      asm volatile("s_wait_asynccnt 0x0" ::: "memory");
    }
    __syncthreads();                      // B tile visible to all waves

    if (k0 + 64 < K)                      // prefetch A two k-tiles ahead
      __builtin_prefetch(Ab + (long)(lane & 15) * lda + k0 + 64, 0, 1);

    v16h a = frag_a_load(Ab, lda, lane, k0);
    const hlf* bl = &sBt[ks & 1][0];
#pragma unroll
    for (int i = 0; i < 4; ++i) {
      v16h b = frag_b_nk(bl, 32, 16 * i, lane, 0);
      acc[i] = __builtin_amdgcn_wmma_f32_16x16x32_f16(
          false, a, false, b, (short)0, acc[i], false, false);
    }
    __syncthreads();                      // done reading before next overwrite
  }

  int hi = lane >> 4, c = lane & 15;
#pragma unroll
  for (int i = 0; i < 4; ++i) {
    int n = n0base + 16 * i + c;
    float bv = bias ? bias[n] : 0.0f;
#pragma unroll
    for (int r = 0; r < 8; ++r) {
      int m = m0 + 8 * hi + r;
      float v = acc[i][r] * alpha + bv;
      if (Cf) {
        long idx = ct ? ((long)n * ldcf + m) : ((long)m * ldcf + n);
        Cf[(long)z * sCf + idx] = v;
      }
      if (Ch) {
        long idx = ct ? ((long)n * ldch + m) : ((long)m * ldch + n);
        Ch[(long)z * sCh + idx] = (hlf)v;
      }
    }
  }
}

// ---------------------------------------------------------------------------
// BiLSTM scan: workgroup = 16 batch lanes, 8 waves, h/c in LDS.
// ---------------------------------------------------------------------------
__device__ __forceinline__ float sigmoidf_(float x) { return 1.0f / (1.0f + __expf(-x)); }

__global__ void __launch_bounds__(256)
lstm_scan(const float* xg_f, const float* xg_b,
          const hlf* Whh_f, const hlf* Whh_b, hlf* hcat)
{
  __shared__ hlf   sh[16 * 128];
  __shared__ float sc[16 * 128];
  __shared__ float sg[16 * 512];

  int dir = blockIdx.y;
  int b0  = blockIdx.x * 16;
  const float* xg  = dir ? xg_b : xg_f;
  const hlf*   Whh = dir ? Whh_b : Whh_f;

  int tid  = threadIdx.x;
  int w    = tid >> 5;
  int lane = tid & 31;
  int hi   = lane >> 4, c = lane & 15;

  for (int i = tid; i < 16 * 128; i += 256) { sh[i] = (hlf)0.0f; sc[i] = 0.0f; }
  __syncthreads();

  for (int step = 0; step < 256; ++step) {
    int t = dir ? (255 - step) : step;

    v8f acc[4] = {};
#pragma unroll
    for (int ks = 0; ks < 4; ++ks) {
      v16h a = frag_a_load(sh, 128, lane, ks * 32);
#pragma unroll
      for (int i = 0; i < 4; ++i) {
        int n0 = w * 64 + 16 * i;
        v16h b = frag_b_nk(Whh, 128, n0, lane, ks * 32);
        acc[i] = __builtin_amdgcn_wmma_f32_16x16x32_f16(
            false, a, false, b, (short)0, acc[i], false, false);
      }
    }
#pragma unroll
    for (int i = 0; i < 4; ++i) {
      int n = w * 64 + 16 * i + c;
#pragma unroll
      for (int r = 0; r < 8; ++r) {
        int m = 8 * hi + r;
        sg[m * 512 + n] = acc[i][r] + xg[((long)(b0 + m) * 256 + t) * 512 + n];
      }
    }
    __syncthreads();

    for (int idx = tid; idx < 16 * 128; idx += 256) {
      int m = idx >> 7, j = idx & 127;
      float ig = sg[m * 512 + j];
      float fg = sg[m * 512 + 128 + j];
      float gg = sg[m * 512 + 256 + j];
      float og = sg[m * 512 + 384 + j];
      float cn = sigmoidf_(fg) * sc[idx] + sigmoidf_(ig) * tanhf(gg);
      float hn = sigmoidf_(og) * tanhf(cn);
      sc[idx] = cn;
      sh[idx] = (hlf)hn;
      hcat[((long)(b0 + m) * 256 + t) * 256 + dir * 128 + j] = (hlf)hn;
    }
    __syncthreads();
  }
}

// ---------------------------------------------------------------------------
// Small support kernels
// ---------------------------------------------------------------------------
__global__ void __launch_bounds__(256)
bn_stats(const float* x, const float* gamma, const float* beta,
         float* scale, float* shift)
{
  __shared__ float ssum[256], ssq[256];
  int col = blockIdx.x, tid = threadIdx.x;
  float s = 0.0f, q = 0.0f;
  for (int r = tid; r < 32768; r += 256) {
    float v = x[(long)r * 768 + col];
    s += v; q += v * v;
  }
  ssum[tid] = s; ssq[tid] = q; __syncthreads();
  for (int o = 128; o; o >>= 1) {
    if (tid < o) { ssum[tid] += ssum[tid + o]; ssq[tid] += ssq[tid + o]; }
    __syncthreads();
  }
  if (tid == 0) {
    float mean = ssum[0] * (1.0f / 32768.0f);
    float var  = ssq[0] * (1.0f / 32768.0f) - mean * mean;
    float sc   = gamma[col] * rsqrtf(var + 1e-5f);
    scale[col] = sc;
    shift[col] = beta[col] - mean * sc;
  }
}

__global__ void __launch_bounds__(256)
bn_apply(const float* x, const float* scale, const float* shift, hlf* xn)
{
  long i = (long)blockIdx.x * 256 + threadIdx.x;
  int  c = (int)(i % 768);
  xn[i] = (hlf)(x[i] * scale[c] + shift[c]);
}

__global__ void __launch_bounds__(256)
cvt_h(const float* in, hlf* out, int n)
{
  int i = blockIdx.x * 256 + threadIdx.x;
  if (i < n) out[i] = (hlf)in[i];
}

__global__ void __launch_bounds__(256)
transpose_h(const float* in, hlf* out, int R, int C)   // out[C][R] = in[R][C]
{
  int i = blockIdx.x * 256 + threadIdx.x;
  if (i < R * C) {
    int r = i / C, c = i % C;
    out[(long)c * R + r] = (hlf)in[i];
  }
}

__global__ void __launch_bounds__(256)
softmax_row(const float* Sc, hlf* P)   // row length 256
{
  __shared__ float red[256];
  long row = blockIdx.x;
  int tid  = threadIdx.x;
  float v = Sc[row * 256 + tid];
  red[tid] = v; __syncthreads();
  for (int o = 128; o; o >>= 1) { if (tid < o) red[tid] = fmaxf(red[tid], red[tid + o]); __syncthreads(); }
  float mx = red[0]; __syncthreads();
  float e = __expf(v - mx);
  red[tid] = e; __syncthreads();
  for (int o = 128; o; o >>= 1) { if (tid < o) red[tid] += red[tid + o]; __syncthreads(); }
  P[row * 256 + tid] = (hlf)(e / red[0]);
}

__global__ void __launch_bounds__(256)
head_mean(const hlf* attn, float* Am)   // Am[b,i,j] = relu(mean_h attn)
{
  long i = (long)blockIdx.x * 256 + threadIdx.x;
  long b = i >> 16;
  long rem = i & 65535;
  float s = 0.0f;
#pragma unroll
  for (int h = 0; h < 4; ++h) s += (float)attn[((b * 4 + h) << 16) + rem];
  Am[i] = fmaxf(0.25f * s, 0.0f);
}

__global__ void __launch_bounds__(256)
colsum_dinv(const float* Am, float* dinv)   // deg[j]=sum_i(A+I)[i,j]
{
  int b = blockIdx.x, j = threadIdx.x;
  float s = 1.0f;
  for (int i = 0; i < 256; ++i) s += Am[((long)b << 16) + (long)i * 256 + j];
  dinv[b * 256 + j] = (s > 0.0f) ? rsqrtf(s) : 0.0f;
}

__global__ void __launch_bounds__(256)
make_ahatT(const float* Am, const float* dinv, hlf* At)  // At[b][j][i] = Ahat[b][i][j]
{
  int bj = blockIdx.x;
  int b = bj >> 8, j = bj & 255;
  int i = threadIdx.x;
  float v = Am[((long)b << 16) + (long)i * 256 + j] + ((i == j) ? 1.0f : 0.0f);
  At[((long)b << 16) + (long)j * 256 + i] =
      (hlf)(dinv[b * 256 + i] * v * dinv[b * 256 + j]);
}

// ---------------------------------------------------------------------------
// Host orchestration
// ---------------------------------------------------------------------------
extern "C" void kernel_launch(void* const* d_in, const int* in_sizes, int n_in,
                              void* d_out, int out_size, void* d_ws, size_t ws_size,
                              hipStream_t stream) {
  (void)in_sizes; (void)n_in; (void)out_size; (void)ws_size;
  const float* x       = (const float*)d_in[0];
  const float* gamma   = (const float*)d_in[1];
  const float* beta    = (const float*)d_in[2];
  const float* WihF    = (const float*)d_in[3];
  const float* WhhF    = (const float*)d_in[4];
  const float* bF      = (const float*)d_in[5];
  const float* WihB    = (const float*)d_in[6];
  const float* WhhB    = (const float*)d_in[7];
  const float* bB      = (const float*)d_in[8];
  const float* glinW   = (const float*)d_in[9];
  const float* glinB   = (const float*)d_in[10];
  const float* edgeW   = (const float*)d_in[11];
  const float* edgeB   = (const float*)d_in[12];
  const float* mhaWin  = (const float*)d_in[13];
  const float* mhaBin  = (const float*)d_in[14];
  const float* mhaWout = (const float*)d_in[15];
  const float* mhaBout = (const float*)d_in[16];
  const float* gcnW1   = (const float*)d_in[17];
  const float* gcnB1   = (const float*)d_in[18];
  const float* gcnW2   = (const float*)d_in[19];
  const float* gcnB2   = (const float*)d_in[20];
  float* dout = (float*)d_out;

  char* ws = (char*)d_ws;
  size_t off = 0;
  auto carve = [&](size_t bytes) -> char* {
    off = (off + 255) & ~(size_t)255;
    char* p = ws + off; off += bytes; return p;
  };

  hlf* wWihF = (hlf*)carve((size_t)512 * 768 * 2);
  hlf* wWihB = (hlf*)carve((size_t)512 * 768 * 2);
  hlf* wWhhF = (hlf*)carve((size_t)512 * 128 * 2);
  hlf* wWhhB = (hlf*)carve((size_t)512 * 128 * 2);
  hlf* wGlin = (hlf*)carve((size_t)128 * 256 * 2);
  hlf* wEdge = (hlf*)carve((size_t)128 * 768 * 2);
  hlf* wMin  = (hlf*)carve((size_t)768 * 256 * 2);
  hlf* wMout = (hlf*)carve((size_t)256 * 256 * 2);
  hlf* wW1t  = (hlf*)carve((size_t)512 * 768 * 2);   // [512,768]
  hlf* wW2t  = (hlf*)carve((size_t)256 * 512 * 2);   // [256,512]
  float* bnScale = (float*)carve(768 * 4);
  float* bnShift = (float*)carve(768 * 4);

  hlf*   xn    = (hlf*)carve(BS * 768 * 2);                    // 48MB
  float* xg_f  = (float*)carve(BS * 512 * 4);                  // 64MB
  float* xg_b  = (float*)carve(BS * 512 * 4);                  // 64MB
  hlf*   hcat  = (hlf*)carve(BS * 256 * 2);                    // 16MB
  hlf*   x1    = (hlf*)carve(BS * 256 * 2);                    // 16MB
  hlf*   qk    = (hlf*)carve(BS * 512 * 2);                    // 32MB
  hlf*   vT    = (hlf*)carve(BS * 256 * 2);                    // 16MB
  hlf*   attn  = (hlf*)carve((size_t)B_ * 4 * 256 * 256 * 2);  // 64MB
  hlf*   ctx   = (hlf*)carve(BS * 256 * 2);                    // 16MB
  float* dinv  = (float*)carve((size_t)B_ * 256 * 4);
  hlf*   xw1T  = (hlf*)carve(BS * 512 * 2);                    // 32MB

  // stream-ordered aliases
  float* scores = xg_f;                          // after LSTM consumed xg
  float* Amean  = (float*)xn;                    // after xn consumed
  hlf*   AhatT  = (hlf*)((char*)xn + (size_t)33554432);
  hlf*   h1     = qk;                            // after scores consumed qk
  hlf*   hw2T   = vT;                            // after ctx consumed vT

  auto gemm_lds = [&](const hlf* A, long sA, int lda,
                      const hlf* Bm, long sB, int ldb,
                      const float* bias, float alpha,
                      float* Cf, long sCf, int ldcf,
                      hlf* Ch, long sCh, int ldch,
                      int M, int N, int K, int Z, int ct) {
    dim3 g(N / 64, M / 128, Z);
    wmma_gemm_lds<<<g, dim3(256), 0, stream>>>(A, sA, lda, Bm, sB, ldb, bias, alpha,
                                               Cf, sCf, ldcf, Ch, sCh, ldch, M, N, K, ct);
  };

  // --- weight conversion ---
  cvt_h<<<(512 * 768 + 255) / 256, 256, 0, stream>>>(WihF, wWihF, 512 * 768);
  cvt_h<<<(512 * 768 + 255) / 256, 256, 0, stream>>>(WihB, wWihB, 512 * 768);
  cvt_h<<<(512 * 128 + 255) / 256, 256, 0, stream>>>(WhhF, wWhhF, 512 * 128);
  cvt_h<<<(512 * 128 + 255) / 256, 256, 0, stream>>>(WhhB, wWhhB, 512 * 128);
  cvt_h<<<(128 * 256 + 255) / 256, 256, 0, stream>>>(glinW, wGlin, 128 * 256);
  cvt_h<<<(128 * 768 + 255) / 256, 256, 0, stream>>>(edgeW, wEdge, 128 * 768);
  cvt_h<<<(768 * 256 + 255) / 256, 256, 0, stream>>>(mhaWin, wMin, 768 * 256);
  cvt_h<<<(256 * 256 + 255) / 256, 256, 0, stream>>>(mhaWout, wMout, 256 * 256);
  transpose_h<<<(768 * 512 + 255) / 256, 256, 0, stream>>>(gcnW1, wW1t, 768, 512);
  transpose_h<<<(512 * 256 + 255) / 256, 256, 0, stream>>>(gcnW2, wW2t, 512, 256);

  // --- BatchNorm ---
  bn_stats<<<768, 256, 0, stream>>>(x, gamma, beta, bnScale, bnShift);
  bn_apply<<<(int)(BS * 768 / 256), 256, 0, stream>>>(x, bnScale, bnShift, xn);

  // --- GEMMs on xn ---
  gemm_lds(xn, 0, 768, wWihF, 0, 768, bF, 1.0f,
           xg_f, 0, 512, nullptr, 0, 0, (int)BS, 512, 768, 1, 0);
  gemm_lds(xn, 0, 768, wWihB, 0, 768, bB, 1.0f,
           xg_b, 0, 512, nullptr, 0, 0, (int)BS, 512, 768, 1, 0);
  gemm_lds(xn, 0, 768, wEdge, 0, 768, edgeB, 1.0f,
           nullptr, 0, 0, x1, 0, 256, (int)BS, 128, 768, 1, 0);
  // xw1T[b][f][i] = (xn W1)[b,i,f]  (batched, transposed write)
  gemm_lds(xn, 196608, 768, wW1t, 0, 768, nullptr, 1.0f,
           nullptr, 0, 0, xw1T, 131072, 256, 256, 512, 768, B_, 1);

  // --- BiLSTM scan ---
  lstm_scan<<<dim3(8, 2), 256, 0, stream>>>(xg_f, xg_b, wWhhF, wWhhB, hcat);

  // --- glinear -> second half of x1 ---
  gemm_lds(hcat, 0, 256, wGlin, 0, 256, glinB, 1.0f,
           nullptr, 0, 0, x1 + 128, 0, 256, (int)BS, 128, 256, 1, 0);

  // --- MHA in-projection: q,k rows -> qk ; v -> vT (transposed per batch) ---
  gemm_lds(x1, 0, 256, wMin, 0, 256, mhaBin, 1.0f,
           nullptr, 0, 0, qk, 0, 512, (int)BS, 512, 256, 1, 0);
  gemm_lds(x1, 65536, 256, wMin + (long)512 * 256, 0, 256, mhaBin + 512, 1.0f,
           nullptr, 0, 0, vT, 65536, 256, 256, 256, 256, B_, 1);

  // --- scores per head: q @ k^T * 1/8 ---
  for (int h = 0; h < 4; ++h) {
    gemm_lds(qk + h * 64, 131072, 512,
             qk + 256 + h * 64, 131072, 512,
             nullptr, 0.125f,
             scores + (long)h * 65536, 262144, 256,
             nullptr, 0, 0,
             256, 256, 64, B_, 0);
  }
  softmax_row<<<(int)((long)B_ * 4 * 256), 256, 0, stream>>>(scores, attn);

  // --- context per head: attn @ v (vT is [N,K]) ---
  for (int h = 0; h < 4; ++h) {
    gemm_lds(attn + (long)h * 65536, 262144, 256,
             vT + (long)h * 64 * 256, 65536, 256,
             nullptr, 1.0f,
             nullptr, 0, 0,
             ctx + h * 64, 65536, 256,
             256, 64, 256, B_, 0);
  }

  // --- out-projection -> lu (second output) ---
  gemm_lds(ctx, 0, 256, wMout, 0, 256, mhaBout, 1.0f,
           dout + (long)B_ * S_ * H_, 0, 256, nullptr, 0, 0, (int)BS, 256, 256, 1, 0);

  // --- dense GCN normalization ---
  head_mean<<<(int)((long)B_ * 256 * 256 / 256), 256, 0, stream>>>(attn, Amean);
  colsum_dinv<<<B_, 256, 0, stream>>>(Amean, dinv);
  make_ahatT<<<B_ * 256, 256, 0, stream>>>(Amean, dinv, AhatT);

  // --- GCN layer 1: h1[b][j][f] = sum_i AhatT[b][j][i] * xw1T[b][f][i] + b1 ---
  gemm_lds(AhatT, 65536, 256, xw1T, 131072, 256, gcnB1, 1.0f,
           nullptr, 0, 0, h1, 131072, 512, 256, 512, 256, B_, 0);
  // --- hw2T[b][f'][i] = (h1 W2)[b,i,f']  (batched, transposed write) ---
  gemm_lds(h1, 131072, 512, wW2t, 0, 512, nullptr, 1.0f,
           nullptr, 0, 0, hw2T, 65536, 256, 256, 256, 512, B_, 1);
  // --- GCN layer 2: h2 = Ahat^T @ hw2 + b2 -> first output ---
  gemm_lds(AhatT, 65536, 256, hw2T, 65536, 256, gcnB2, 1.0f,
           dout, 65536, 256, nullptr, 0, 0, 256, 256, 256, B_, 0);
}
